// HistEncoder_25048249270530
// MI455X (gfx1250) — compile-verified, matrix operand order checked by reading, not run on hardware
//
#include <hip/hip_runtime.h>
#include <hip/hip_bf16.h>

// ---------------------------------------------------------------------------
// Model constants (from reference)
// ---------------------------------------------------------------------------
#define TT    32      // T
#define BB    32      // B
#define FEAT_ 4
#define DD    256     // D
#define HH    8       // heads (encoder / grid attn)
#define HD_   32      // head dim
#define DFF_  1024
#define CC    169     // G*G
#define NN    1024    // neighbor sequences
#define HID2  512     // HID
#define SEQ_  1056    // BB + NN sequences through shared encoder
#define MTOK  33792   // SEQ_*TT tokens
#define NEGV  (-1.0e9f)

typedef __attribute__((ext_vector_type(16))) _Float16 v16h;
typedef __attribute__((ext_vector_type(8)))  float    v8f;
typedef __attribute__((ext_vector_type(2)))  _Float16 h2v;

// ---------------------------------------------------------------------------
// Device helpers
// ---------------------------------------------------------------------------
__device__ __forceinline__ float actf(float v, int act) {
  switch (act) {
    case 1: return v > 0.f ? v : 0.f;                                  // relu
    case 2: return 1.f / (1.f + __expf(-v));                           // sigmoid
    case 3: {                                                          // gelu(tanh)
      float t = 0.7978845608028654f * (v + 0.044715f * v * v * v);
      return 0.5f * v * (1.f + tanhf(t));
    }
    case 4: return v / (1.f + __expf(-v));                             // silu
    default: return v;
  }
}

__device__ __forceinline__ float blockReduceSum256(float v) {
  __shared__ float sh[8];
  for (int o = 16; o; o >>= 1) v += __shfl_xor(v, o, 32);
  int w = threadIdx.x >> 5, l = threadIdx.x & 31;
  if (l == 0) sh[w] = v;
  __syncthreads();
  if (w == 0) {
    v = (l < 8) ? sh[l] : 0.f;
    for (int o = 4; o; o >>= 1) v += __shfl_xor(v, o, 32);
    if (l == 0) sh[0] = v;
  }
  __syncthreads();
  v = sh[0];
  __syncthreads();
  return v;
}

// ---------------------------------------------------------------------------
// WMMA GEMM:  out[M,N] = act(A[M,K] * W[N,K]^T + bias[N])
// A,W f16; out f32 and/or f16. M % 16 == 0, N % 32 == 0, K % 32 == 0.
// Block = 8 waves; wave tile 16x16; block tile 64 (M) x 32 (N).
// A-fragment layout per CDNA5 ISA 7.12.2 (16-bit A 16x32); B mirrors it with
// W stored row-major N x K (so lane n reads W[n][k] == B[k][n]).
// ---------------------------------------------------------------------------
__global__ __launch_bounds__(256)
void k_wmma_linear(const _Float16* __restrict__ A, const _Float16* __restrict__ W,
                   const float* __restrict__ bias,
                   float* __restrict__ outF, _Float16* __restrict__ outH,
                   int M, int N, int K, int act)
{
  const int wave = threadIdx.x >> 5;
  const int lane = threadIdx.x & 31;
  const int m0 = blockIdx.y * 64 + (wave & 3) * 16;
  const int n0 = blockIdx.x * 32 + (wave >> 2) * 16;
  if (m0 >= M || n0 >= N) return;          // wave-uniform; EXEC stays all-ones
  const int hgrp = lane >> 4;
  const int r16  = lane & 15;
  const _Float16* Arow = A + (long)(m0 + r16) * K;
  const _Float16* Wrow = W + (long)(n0 + r16) * K;
  v8f c = {0.f, 0.f, 0.f, 0.f, 0.f, 0.f, 0.f, 0.f};
  for (int kb = 0; kb < K; kb += 32) {
    if (kb + 64 <= K) {                    // global_prefetch_b8 of next K-slab
      __builtin_prefetch(Arow + kb + 32, 0, 0);
      __builtin_prefetch(Wrow + kb + 32, 0, 0);
    }
    v16h a, b;
#pragma unroll
    for (int p = 0; p < 8; ++p) {
      const int kk = kb + 8 * hgrp + ((2 * p) & 7) + ((p >> 2) << 4);
      h2v av = *(const h2v*)(Arow + kk);
      h2v bv = *(const h2v*)(Wrow + kk);
      a[2 * p] = av[0]; a[2 * p + 1] = av[1];
      b[2 * p] = bv[0]; b[2 * p + 1] = bv[1];
    }
    c = __builtin_amdgcn_wmma_f32_16x16x32_f16(false, a, false, b, (short)0, c,
                                               false, false);
  }
  const int n  = n0 + r16;
  const int mb = m0 + hgrp * 8;
  const float bv = bias ? bias[n] : 0.f;
#pragma unroll
  for (int rr = 0; rr < 8; ++rr) {
    float v = actf(c[rr] + bv, act);
    const long o = (long)(mb + rr) * N + n;
    if (outF) outF[o] = v;
    if (outH) outH[o] = (_Float16)v;
  }
}

// ---------------------------------------------------------------------------
// Small-K (K=4) linear for FEAT->D projections. out[M,N]=A[M,4]*W[N,4]^T+b
// ---------------------------------------------------------------------------
__global__ void k_smallk(const float* __restrict__ A, const float* __restrict__ W,
                         const float* __restrict__ bias, float* __restrict__ outF,
                         long M, int N)
{
  long i = blockIdx.x * 256L + threadIdx.x;
  if (i >= M * N) return;
  long m = i / N; int n = (int)(i % N);
  const float* a = A + m * FEAT_;
  const float* w = W + (long)n * FEAT_;
  float acc = bias[n];
#pragma unroll
  for (int k = 0; k < FEAT_; ++k) acc += a[k] * w[k];
  outF[i] = acc;
}

// ---------------------------------------------------------------------------
// Elementwise + conversion kernels
// ---------------------------------------------------------------------------
__global__ void k_cvt(const float* a, _Float16* o, long n) {
  long i = blockIdx.x * 256L + threadIdx.x;
  if (i < n) o[i] = (_Float16)a[i];
}
__global__ void k_addcvt(const float* a, const float* b, _Float16* o, long n) {
  long i = blockIdx.x * 256L + threadIdx.x;
  if (i < n) o[i] = (_Float16)(a[i] + b[i]);
}
// mode 0: copy a; 1: a+b; 2: leaky(a+b); 3: a*b; 4: leaky(a)
__global__ void k_elt(const float* a, const float* b, float* o, long n, int mode) {
  long i = blockIdx.x * 256L + threadIdx.x;
  if (i >= n) return;
  float v;
  switch (mode) {
    case 1: v = a[i] + b[i]; break;
    case 2: { float s = a[i] + b[i]; v = s > 0.f ? s : 0.1f * s; } break;
    case 3: v = a[i] * b[i]; break;
    case 4: { float s = a[i]; v = s > 0.f ? s : 0.1f * s; } break;
    default: v = a[i];
  }
  o[i] = v;
}
__global__ void k_concat2(const float* a, const float* b, float* o, long rows, int d) {
  long i = blockIdx.x * 256L + threadIdx.x;
  if (i >= rows * 2 * d) return;
  long r = i / (2 * d); int col = (int)(i % (2 * d));
  o[i] = (col < d) ? a[r * d + col] : b[r * d + (col - d)];
}

// ---------------------------------------------------------------------------
// Positional encoding (per token from its own features, per reference)
// ---------------------------------------------------------------------------
__global__ void k_pos_enc(const float* __restrict__ feats, float* __restrict__ pos,
                          long rows)
{
  long i = blockIdx.x * 256L + threadIdx.x;
  if (i >= rows * DD) return;
  long r = i >> 8; int d = (int)(i & 255);
  float e = feats[r * FEAT_ + (d < 128 ? 1 : 0)];
  int dd = d & 127;
  int pair = dd >> 1;
  float freq = __powf(10000.f, -(float)(2 * pair) / 128.f);
  float ang = e * 6.283185307179586f * freq;
  pos[i] = (dd & 1) ? __cosf(ang) : __sinf(ang);
}

// ---------------------------------------------------------------------------
// LayerNorm over last dim; optional residual; optional f32/f16 outputs.
// One block (256 threads) per row.
// ---------------------------------------------------------------------------
__global__ __launch_bounds__(256)
void k_layernorm(const float* __restrict__ in, const float* __restrict__ res,
                 const float* __restrict__ g, const float* __restrict__ b,
                 float* __restrict__ out, _Float16* __restrict__ outH, int dim)
{
  long row = blockIdx.x;
  const float* p  = in + row * (long)dim;
  const float* rp = res ? res + row * (long)dim : nullptr;
  float s = 0.f;
  for (int d = threadIdx.x; d < dim; d += 256) s += p[d] + (rp ? rp[d] : 0.f);
  float mean = blockReduceSum256(s) / dim;
  float s2 = 0.f;
  for (int d = threadIdx.x; d < dim; d += 256) {
    float v = p[d] + (rp ? rp[d] : 0.f) - mean; s2 += v * v;
  }
  float inv = rsqrtf(blockReduceSum256(s2) / dim + 1e-5f);
  for (int d = threadIdx.x; d < dim; d += 256) {
    float v = (p[d] + (rp ? rp[d] : 0.f) - mean) * inv * g[d] + b[d];
    if (out)  out[row * (long)dim + d]  = v;
    if (outH) outH[row * (long)dim + d] = (_Float16)v;
  }
}

// ---------------------------------------------------------------------------
// Encoder self-attention: per sequence of T=32 tokens, H=8 heads, HD=32.
// Block 256 = 8 waves; wave == head; lane == key index / out dim.
// q,k,v,out f16, rows = s*T + t, col = h*32 + d.
// ---------------------------------------------------------------------------
__global__ __launch_bounds__(256)
void k_enc_attn(const _Float16* __restrict__ q, const _Float16* __restrict__ k,
                const _Float16* __restrict__ v, _Float16* __restrict__ out)
{
  int s = blockIdx.x;
  int hh = threadIdx.x >> 5;
  int lane = threadIdx.x & 31;
  long base = (long)s * TT * DD + hh * HD_;
  float kreg[HD_];
#pragma unroll
  for (int d = 0; d < HD_; ++d) kreg[d] = (float)k[base + (long)lane * DD + d];
  __shared__ float wsh[HH][TT];
  for (int t = 0; t < TT; ++t) {
    float acc = 0.f;
#pragma unroll
    for (int d = 0; d < HD_; ++d) acc += (float)q[base + (long)t * DD + d] * kreg[d];
    float sc = acc * 0.17677669529663687f;  // 1/sqrt(32)
    float mx = sc;
    for (int o = 16; o; o >>= 1) mx = fmaxf(mx, __shfl_xor(mx, o, 32));
    float w = __expf(sc - mx);
    float sum = w;
    for (int o = 16; o; o >>= 1) sum += __shfl_xor(sum, o, 32);
    wsh[hh][lane] = w / sum;
    __syncthreads();
    float o2 = 0.f;
    for (int j = 0; j < TT; ++j)
      o2 += wsh[hh][j] * (float)v[base + (long)j * DD + lane];
    out[base + (long)t * DD + lane] = (_Float16)o2;
    __syncthreads();
  }
}

// ---------------------------------------------------------------------------
// Occupancy / scatter-index helpers
// ---------------------------------------------------------------------------
__global__ void k_any(const int* __restrict__ m, int* __restrict__ occ,
                      int per, int ncells)
{
  int c = blockIdx.x * 256 + threadIdx.x;
  if (c >= ncells) return;
  const int* p = m + (long)c * per;
  int r = 0;
  for (int i = 0; i < per; ++i) r |= (p[i] != 0);
  occ[c] = r;
}
__global__ void k_scan(const int* occ, int* idx, int n, int mx) {
  if (threadIdx.x || blockIdx.x) return;
  int cum = 0;
  for (int i = 0; i < n; ++i) {
    cum += (occ[i] != 0);
    int v = cum - 1; v = v < 0 ? 0 : (v > mx ? mx : v);
    idx[i] = v;
  }
}

// ---------------------------------------------------------------------------
// Grid attention (shared by temporal & social branch).
// q: (b*T+t, 256). kall/vall: (n*T+t, 256), gathered via idx. out: (b*T+t,256)
// Block 256 per (b,t); wave == head.
// ---------------------------------------------------------------------------
__global__ __launch_bounds__(256)
void k_grid_attn(const float* __restrict__ q, const float* __restrict__ kall,
                 const float* __restrict__ vall, const int* __restrict__ occ,
                 const int* __restrict__ idx, float* __restrict__ out)
{
  int bt = blockIdx.x;
  int b = bt / TT, t = bt % TT;
  int hh = threadIdx.x >> 5, lane = threadIdx.x & 31;
  __shared__ float wsh[HH][176];
  float qreg[HD_];
  const float* qp = q + (long)bt * DD + hh * HD_;
#pragma unroll
  for (int d = 0; d < HD_; ++d) qreg[d] = qp[d];
  float e[6];
  float mx = -1e30f;
  for (int ch = 0; ch < 6; ++ch) {
    int c = ch * 32 + lane;
    float s = -1e30f;
    if (c < CC && occ[b * CC + c]) {
      const float* kp = kall + ((long)idx[b * CC + c] * TT + t) * DD + hh * HD_;
      float acc = 0.f;
#pragma unroll
      for (int d = 0; d < HD_; ++d) acc += qreg[d] * kp[d];
      s = acc * 0.17677669529663687f;
    }
    e[ch] = s;
    mx = fmaxf(mx, s);
  }
  for (int o = 16; o; o >>= 1) mx = fmaxf(mx, __shfl_xor(mx, o, 32));
  float sum = 0.f;
  for (int ch = 0; ch < 6; ++ch) {
    float w = (e[ch] > -1e29f) ? __expf(e[ch] - mx) : 0.f;
    e[ch] = w; sum += w;
  }
  for (int o = 16; o; o >>= 1) sum += __shfl_xor(sum, o, 32);
  float inv = 1.f / (sum + 1e-6f);
  for (int ch = 0; ch < 6; ++ch) {
    int c = ch * 32 + lane;
    if (c < 176) wsh[hh][c] = e[ch] * inv;
  }
  __syncthreads();
  float acc = 0.f;
  for (int c = 0; c < CC; ++c) {
    float w = wsh[hh][c];
    if (w != 0.f) {
      const float* vp = vall + ((long)idx[b * CC + c] * TT + t) * DD + hh * HD_;
      acc += w * vp[lane];
    }
  }
  out[(long)bt * DD + hh * HD_ + lane] = acc;
}

// ---------------------------------------------------------------------------
// Neighbor scoring / top-k / summaries (x = full encoder output, f32)
// ---------------------------------------------------------------------------
__global__ __launch_bounds__(256)
void k_score(const float* __restrict__ x, const int* __restrict__ occ,
             const int* __restrict__ idx, float* __restrict__ score)
{
  int cell = blockIdx.x;             // b*169 + c
  long base = (long)(BB + idx[cell]) * TT;
  float s = 0.f;
  for (int i = threadIdx.x; i < TT * DD; i += 256) {
    int t = i >> 8, d = i & 255;
    float v = x[(base + t) * DD + d];
    s += v * v;
  }
  s = blockReduceSum256(s);
  if (threadIdx.x == 0)
    score[cell] = occ[cell] ? s * (1.f / (TT * DD)) : -__builtin_inff();
}
__global__ void k_topk(const float* __restrict__ score, int* __restrict__ topi,
                       float* __restrict__ validf)
{
  int b = threadIdx.x;
  if (b >= BB || blockIdx.x) return;
  int chosen[4];
  for (int j = 0; j < 4; ++j) {
    float best = -__builtin_inff(); int bi = 0; bool found = false;
    for (int c = 0; c < CC; ++c) {
      bool used = false;
      for (int p = 0; p < j; ++p) if (chosen[p] == c) used = true;
      if (used) continue;
      float s = score[b * CC + c];
      if (!found || s > best) { best = s; bi = c; found = true; }
    }
    chosen[j] = bi;
    topi[b * 4 + j] = bi;
    validf[b * 4 + j] = (best > -1e30f) ? 1.f : 0.f;
  }
}
__global__ void k_summ(const float* __restrict__ x, const int* __restrict__ idx,
                       const int* __restrict__ topi, const float* __restrict__ validf,
                       float* __restrict__ summ)
{
  int bj = blockIdx.x;               // b*4 + j
  int d = threadIdx.x;
  int b = bj >> 2;
  float v = 0.f;
  if (validf[bj] != 0.f) {
    long base = (long)(BB + idx[b * CC + topi[bj]]) * TT;
    for (int t = 0; t < TT; ++t) v += x[(base + t) * DD + d];
    v *= (1.f / TT);
  }
  summ[(long)bj * DD + d] = v;
}

// ---------------------------------------------------------------------------
// Memory-token assembly, mask, intent-query broadcast
// ---------------------------------------------------------------------------
__global__ void k_mt(const float* fh, const float* nh, float* mt, float* dout) {
  long i = blockIdx.x * 256L + threadIdx.x;
  if (i >= 1152L * HID2) return;
  int row = (int)(i >> 9), col = (int)(i & 511);
  int b = row / 36, j = row % 36;
  float v = (j < TT) ? fh[((long)b * TT + j) * HID2 + col]
                     : nh[((long)b * 4 + (j - TT)) * HID2 + col];
  mt[i] = v; dout[i] = v;
}
__global__ void k_mask(const float* validf, float* maskf, float* dout) {
  int i = blockIdx.x * 256 + threadIdx.x;
  if (i >= 1152) return;
  int b = i / 36, j = i % 36;
  float v = (j < TT) ? 0.f : (1.f - validf[b * 4 + (j - TT)]);
  maskf[i] = v; dout[i] = v;
}
__global__ void k_iq(const float* cls, float* iqF, _Float16* iqH) {
  int i = blockIdx.x * 256 + threadIdx.x;
  if (i >= 64 * HID2) return;
  int row = i >> 9, qi = row & 1;
  float v = cls[(qi << 9) + (i & 511)];
  iqF[i] = v; iqH[i] = (_Float16)v;
}

// ---------------------------------------------------------------------------
// Intent attention: b=32, 2 queries, 36 keys, 4 heads of 128.
// Block 128 per (b,qi); wave == head; lane covers 4 strided dims.
// ---------------------------------------------------------------------------
__global__ __launch_bounds__(128)
void k_intent_attn(const _Float16* __restrict__ q, const _Float16* __restrict__ k,
                   const _Float16* __restrict__ v, const float* __restrict__ maskf,
                   float* __restrict__ out)
{
  int bq = blockIdx.x;
  int b = bq >> 1;
  int hh = threadIdx.x >> 5, lane = threadIdx.x & 31;
  float qreg[4];
  const _Float16* qp = q + (long)bq * HID2 + hh * 128;
#pragma unroll
  for (int ii = 0; ii < 4; ++ii) qreg[ii] = (float)qp[lane + 32 * ii];
  float sc[36];
  float mx = -1e30f;
  for (int j = 0; j < 36; ++j) {
    const _Float16* kp = k + ((long)b * 36 + j) * HID2 + hh * 128;
    float acc = 0.f;
#pragma unroll
    for (int ii = 0; ii < 4; ++ii) acc += qreg[ii] * (float)kp[lane + 32 * ii];
    for (int o = 16; o; o >>= 1) acc += __shfl_xor(acc, o, 32);
    float s = acc * 0.08838834764831845f;   // 1/sqrt(128)
    if (maskf[b * 36 + j] != 0.f) s = NEGV;
    sc[j] = s; mx = fmaxf(mx, s);
  }
  float sum = 0.f;
  for (int j = 0; j < 36; ++j) { float w = __expf(sc[j] - mx); sc[j] = w; sum += w; }
  float inv = 1.f / sum;
#pragma unroll
  for (int ii = 0; ii < 4; ++ii) {
    float acc = 0.f;
    for (int j = 0; j < 36; ++j) {
      const _Float16* vp = v + ((long)b * 36 + j) * HID2 + hh * 128;
      acc += sc[j] * inv * (float)vp[lane + 32 * ii];
    }
    out[(long)bq * HID2 + hh * 128 + lane + 32 * ii] = acc;
  }
}

// ---------------------------------------------------------------------------
// Host orchestration
// ---------------------------------------------------------------------------
extern "C" void kernel_launch(void* const* d_in, const int* in_sizes, int n_in,
                              void* d_out, int out_size, void* d_ws, size_t ws_size,
                              hipStream_t stream)
{
  (void)in_sizes; (void)n_in; (void)out_size; (void)ws_size;
  auto P = [&](int i) { return (const float*)d_in[i]; };
  auto PI = [&](int i) { return (const int*)d_in[i]; };

  // ---- param leaf indices (setup_inputs insertion order) ----
  // 0 src  1 nbrs  2 mask  3 temporal_mask
  // 4/5 input_embedding w,b ; 6/7 tq ; 8/9 tk ; 10/11 tv
  // 12/13 tr.fc w,b ; 14/15 tr.ln g,b ; 16/17 qf ; 18/19 kf ; 20/21 vf
  // enc l: base=22+16l: q w,b | k w,b | v w,b | o w,b | lin1 w,b | lin2 w,b |
  //        ln1 g,b | ln2 g,b
  // 70/71 fg.ln ; 72/73 fg.lin ; 74/75 fusion_norm ; 76/77 f2h.ln ; 78/79 f2h.lin
  // 80/81 n2h.ln ; 82/83 n2h.lin ; 84 intent_cls ; 85/86 intent_mem_norm
  // 87..94 intent_attn q/k/v/o w,b ; 95/96 intent_cls_norm
  // 97/98 if.ln ; 99/100 if.lin1 ; 101/102 if.lin2

  char* wp = (char*)d_ws;
  auto alloc = [&](size_t bytes) {
    void* p = (void*)wp;
    wp += (bytes + 255) & ~(size_t)255;
    return p;
  };
  auto cvt = [&](const float* src, _Float16* dst, long n) {
    k_cvt<<<dim3((unsigned)((n + 255) / 256)), 256, 0, stream>>>(src, dst, n);
  };
  auto cvtw = [&](int idx, long n) {
    _Float16* h = (_Float16*)alloc(n * 2);
    cvt(P(idx), h, n);
    return h;
  };
  auto gemm = [&](const _Float16* A, const _Float16* W, const float* bias,
                  float* oF, _Float16* oH, int M, int N, int K, int act) {
    dim3 g(N / 32, (M + 63) / 64);
    k_wmma_linear<<<g, 256, 0, stream>>>(A, W, bias, oF, oH, M, N, K, act);
  };
  auto ln = [&](const float* in, const float* res, const float* g, const float* b,
                float* out, _Float16* outH, int rows, int dim) {
    k_layernorm<<<rows, 256, 0, stream>>>(in, res, g, b, out, outH, dim);
  };

  // ---- f16 weights ----
  _Float16* trfc_h = cvtw(12, DD * DD);
  _Float16* qf_h = cvtw(16, DD * DD);
  _Float16* kf_h = cvtw(18, DD * DD);
  _Float16* vf_h = cvtw(20, DD * DD);
  _Float16 *wq_h[3], *wk_h[3], *wv_h[3], *wo_h[3], *w1_h[3], *w2_h[3];
  for (int l = 0; l < 3; ++l) {
    int base = 22 + 16 * l;
    wq_h[l] = cvtw(base + 0, DD * DD);
    wk_h[l] = cvtw(base + 2, DD * DD);
    wv_h[l] = cvtw(base + 4, DD * DD);
    wo_h[l] = cvtw(base + 6, DD * DD);
    w1_h[l] = cvtw(base + 8, (long)DFF_ * DD);
    w2_h[l] = cvtw(base + 10, (long)DD * DFF_);
  }
  _Float16* fg_h  = cvtw(72, (long)DD * 2 * DD);
  _Float16* f2h_h = cvtw(78, (long)HID2 * DD);
  _Float16* n2h_h = cvtw(82, (long)HID2 * DD);
  _Float16* iwq_h = cvtw(87, (long)HID2 * HID2);
  _Float16* iwk_h = cvtw(89, (long)HID2 * HID2);
  _Float16* iwv_h = cvtw(91, (long)HID2 * HID2);
  _Float16* iwo_h = cvtw(93, (long)HID2 * HID2);
  _Float16* if1_h = cvtw(99, (long)HID2 * 2 * HID2);
  _Float16* if2_h = cvtw(101, (long)HID2 * HID2);

  // ---- big activation buffers (ordering matters for post-encoder aliasing) ----
  float*    xbuf  = (float*)alloc((long)MTOK * DD * 4);
  _Float16* x_h   = (_Float16*)alloc((long)MTOK * DD * 2);
  float*    pos   = (float*)alloc((long)MTOK * DD * 4);
  _Float16* qk_h  = (_Float16*)alloc((long)MTOK * DD * 2);   // also attn out
  _Float16* q_h   = (_Float16*)alloc((long)MTOK * DD * 2);
  _Float16* k_hb  = (_Float16*)alloc((long)MTOK * DD * 2);
  _Float16* v_hb  = (_Float16*)alloc((long)MTOK * DD * 2);
  float*    tmp32 = (float*)alloc((long)MTOK * DD * 4);
  _Float16* ff1_h = (_Float16*)alloc((long)MTOK * DFF_ * 2);
  (void)k_hb;
  // post-encoder aliases (those regions are dead after the encoder):
  float* kfall = (float*)q_h;                 // 32768x256 f32 (q_h+k_h regions)
  float* vfall = (float*)v_hb;                // 32768x256 f32 (v_h+tmp32 head)
  float* tkall = (float*)ff1_h;               // 32768x256 f32
  float* tvall = tkall + (long)NN * TT * DD;  // 32768x256 f32

  // ---- small buffers ----
  float* emb_hist = (float*)alloc((long)BB * TT * DD * 4);
  float* tq    = (float*)alloc((long)BB * TT * DD * 4);
  float* t_out = (float*)alloc((long)BB * TT * DD * 4);
  _Float16* touth = (_Float16*)alloc((long)BB * TT * DD * 2);
  float* tfc   = (float*)alloc((long)BB * TT * DD * 4);
  float* tval  = (float*)alloc((long)BB * TT * DD * 4);
  float* qfb   = (float*)alloc((long)BB * TT * DD * 4);
  float* s_out = (float*)alloc((long)BB * TT * DD * 4);
  float* agg   = (float*)alloc((long)BB * TT * DD * 4);
  float* tmpf  = (float*)alloc((long)BB * TT * DD * 4);
  float* fused = (float*)alloc((long)BB * TT * DD * 4);
  float* catb  = (float*)alloc((long)BB * TT * 2 * DD * 4);
  _Float16* cat_h = (_Float16*)alloc((long)BB * TT * 2 * DD * 2);
  float* gate  = (float*)alloc((long)BB * TT * DD * 4);
  _Float16* fln_h = (_Float16*)alloc((long)BB * TT * DD * 2);
  float* fh    = (float*)alloc((long)BB * TT * HID2 * 4);
  int* occT = (int*)alloc(BB * CC * 4);
  int* occS = (int*)alloc(BB * CC * 4);
  int* idxT = (int*)alloc(BB * CC * 4);
  int* idxS = (int*)alloc(BB * CC * 4);
  float* scoreb = (float*)alloc(BB * CC * 4);
  int* topi = (int*)alloc(BB * 4 * 4);
  float* validf = (float*)alloc(BB * 4 * 4);
  float* summb = (float*)alloc((long)BB * 4 * DD * 4);
  _Float16* sln_h = (_Float16*)alloc((long)BB * 4 * DD * 2);
  float* nh = (float*)alloc((long)BB * 4 * HID2 * 4);
  float* mt = (float*)alloc(1152L * HID2 * 4);
  float* maskf = (float*)alloc(1152 * 4);
  _Float16* memn_h = (_Float16*)alloc(1152L * HID2 * 2);
  float* iqF = (float*)alloc(64L * HID2 * 4);
  _Float16* iqH = (_Float16*)alloc(64L * HID2 * 2);
  _Float16* qih = (_Float16*)alloc(64L * HID2 * 2);
  _Float16* kih = (_Float16*)alloc(1152L * HID2 * 2);
  _Float16* vih = (_Float16*)alloc(1152L * HID2 * 2);
  float* att64 = (float*)alloc(64L * HID2 * 4);
  _Float16* att64h = (_Float16*)alloc(64L * HID2 * 2);
  float* ot = (float*)alloc(64L * HID2 * 4);
  float* itb = (float*)alloc(64L * HID2 * 4);
  _Float16* zh = (_Float16*)alloc(32L * 1024 * 2);
  _Float16* h1h = (_Float16*)alloc(32L * HID2 * 2);

  const long NT256 = (long)MTOK * DD;
  auto B1 = [](long n) { return dim3((unsigned)((n + 255) / 256)); };

  // ---- embeddings + positional encodings (tokens = [hist | nbrs]) ----
  k_smallk<<<B1((long)BB * TT * DD), 256, 0, stream>>>(P(0), P(4), P(5), xbuf,
                                                       (long)BB * TT, DD);
  k_smallk<<<B1((long)NN * TT * DD), 256, 0, stream>>>(P(1), P(4), P(5),
                                                       xbuf + (long)BB * TT * DD,
                                                       (long)NN * TT, DD);
  k_elt<<<B1((long)BB * TT * DD), 256, 0, stream>>>(xbuf, nullptr, emb_hist,
                                                    (long)BB * TT * DD, 0);
  k_pos_enc<<<B1((long)BB * TT * DD), 256, 0, stream>>>(P(0), pos, (long)BB * TT);
  k_pos_enc<<<B1((long)NN * TT * DD), 256, 0, stream>>>(P(1),
                                                        pos + (long)BB * TT * DD,
                                                        (long)NN * TT);
  k_elt<<<B1(NT256), 256, 0, stream>>>(xbuf, nullptr, xbuf, NT256, 4);  // leaky

  // ---- shared encoder (3 layers over 1056 sequences) ----
  for (int l = 0; l < 3; ++l) {
    int base = 22 + 16 * l;
    k_addcvt<<<B1(NT256), 256, 0, stream>>>(xbuf, pos, qk_h, NT256);
    cvt(xbuf, x_h, NT256);
    gemm(qk_h, wq_h[l], P(base + 1), nullptr, q_h, MTOK, DD, DD, 0);
    gemm(qk_h, wk_h[l], P(base + 3), nullptr, k_hb, MTOK, DD, DD, 0);
    gemm(x_h,  wv_h[l], P(base + 5), nullptr, v_hb, MTOK, DD, DD, 0);
    k_enc_attn<<<SEQ_, 256, 0, stream>>>(q_h, k_hb, v_hb, qk_h);
    gemm(qk_h, wo_h[l], P(base + 7), tmp32, nullptr, MTOK, DD, DD, 0);
    ln(tmp32, xbuf, P(base + 12), P(base + 13), xbuf, nullptr, MTOK, DD);
    cvt(xbuf, x_h, NT256);
    gemm(x_h, w1_h[l], P(base + 9), nullptr, ff1_h, MTOK, DFF_, DD, 1);  // relu
    gemm(ff1_h, w2_h[l], P(base + 11), tmp32, nullptr, MTOK, DD, DFF_, 0);
    ln(tmp32, xbuf, P(base + 14), P(base + 15), xbuf, nullptr, MTOK, DD);
  }
  cvt(xbuf, x_h, NT256);  // x_h = final encoder output (hist_enc | nbrs_enc)

  // ---- occupancy + scatter indices ----
  k_any<<<B1(BB * CC), 256, 0, stream>>>(PI(3), occT, FEAT_, BB * CC);
  k_any<<<B1(BB * CC), 256, 0, stream>>>(PI(2), occS, DD, BB * CC);
  k_scan<<<1, 1, 0, stream>>>(occT, idxT, BB * CC, NN - 1);
  k_scan<<<1, 1, 0, stream>>>(occS, idxS, BB * CC, NN - 1);

  // ---- temporal branch (raw-feature projections, gathered grid attention) ----
  k_smallk<<<B1((long)BB * TT * DD), 256, 0, stream>>>(P(0), P(6), P(7), tq,
                                                       (long)BB * TT, DD);
  k_smallk<<<B1((long)NN * TT * DD), 256, 0, stream>>>(P(1), P(8), P(9), tkall,
                                                       (long)NN * TT, DD);
  k_smallk<<<B1((long)NN * TT * DD), 256, 0, stream>>>(P(1), P(10), P(11), tvall,
                                                       (long)NN * TT, DD);
  k_grid_attn<<<BB * TT, 256, 0, stream>>>(tq, tkall, tvall, occT, idxT, t_out);
  cvt(t_out, touth, (long)BB * TT * DD);
  gemm(touth, trfc_h, P(13), tfc, nullptr, BB * TT, DD, DD, 0);
  ln(tfc, emb_hist, P(14), P(15), tval, nullptr, BB * TT, DD);

  // ---- social branch ----
  gemm(x_h, qf_h, P(17), qfb, nullptr, BB * TT, DD, DD, 0);  // hist rows
  gemm(x_h + (long)BB * TT * DD, kf_h, P(19), kfall, nullptr, NN * TT, DD, DD, 0);
  gemm(x_h + (long)BB * TT * DD, vf_h, P(21), vfall, nullptr, NN * TT, DD, DD, 0);
  k_grid_attn<<<BB * TT, 256, 0, stream>>>(qfb, kfall, vfall, occS, idxS, s_out);

  // ---- fusion ----
  k_elt<<<B1((long)BB * TT * DD), 256, 0, stream>>>(tval, s_out, agg,
                                                    (long)BB * TT * DD, 2);
  k_concat2<<<B1((long)BB * TT * 2 * DD), 256, 0, stream>>>(xbuf, agg, catb,
                                                            (long)BB * TT, DD);
  ln(catb, nullptr, P(70), P(71), nullptr, cat_h, BB * TT, 2 * DD);
  gemm(cat_h, fg_h, P(73), gate, nullptr, BB * TT, DD, 2 * DD, 2);  // sigmoid
  k_elt<<<B1((long)BB * TT * DD), 256, 0, stream>>>(gate, agg, tmpf,
                                                    (long)BB * TT * DD, 3);
  ln(tmpf, xbuf, P(74), P(75), fused, nullptr, BB * TT, DD);
  ln(fused, nullptr, P(76), P(77), nullptr, fln_h, BB * TT, DD);
  gemm(fln_h, f2h_h, P(79), fh, nullptr, BB * TT, HID2, DD, 3);  // gelu

  // ---- neighbor summaries ----
  k_score<<<BB * CC, 256, 0, stream>>>(xbuf, occS, idxS, scoreb);
  k_topk<<<1, 32, 0, stream>>>(scoreb, topi, validf);
  k_summ<<<BB * 4, 256, 0, stream>>>(xbuf, idxS, topi, validf, summb);
  ln(summb, nullptr, P(80), P(81), nullptr, sln_h, BB * 4, DD);
  gemm(sln_h, n2h_h, P(83), nh, nullptr, BB * 4, HID2, DD, 3);  // gelu

  // ---- memory tokens + mask (outputs 1 and 3) ----
  float* dout = (float*)d_out;
  const long MT_OFF = 0, Z_OFF = 1152L * HID2, MK_OFF = Z_OFF + 32L * HID2;
  k_mt<<<B1(1152L * HID2), 256, 0, stream>>>(fh, nh, mt, dout + MT_OFF);
  k_mask<<<5, 256, 0, stream>>>(validf, maskf, dout + MK_OFF);

  // ---- intent attention ----
  ln(mt, nullptr, P(85), P(86), nullptr, memn_h, 1152, HID2);
  k_iq<<<128, 256, 0, stream>>>(P(84), iqF, iqH);
  gemm(iqH,    iwq_h, P(88), nullptr, qih, 64, HID2, HID2, 0);
  gemm(memn_h, iwk_h, P(90), nullptr, kih, 1152, HID2, HID2, 0);
  gemm(memn_h, iwv_h, P(92), nullptr, vih, 1152, HID2, HID2, 0);
  k_intent_attn<<<64, 128, 0, stream>>>(qih, kih, vih, maskf, att64);
  cvt(att64, att64h, 64L * HID2);
  gemm(att64h, iwo_h, P(94), ot, nullptr, 64, HID2, HID2, 0);
  ln(ot, iqF, P(95), P(96), itb, nullptr, 64, HID2);

  // ---- intent fusion head (output 2: z) ----
  ln(itb, nullptr, P(97), P(98), nullptr, zh, 32, 2 * HID2);   // itb == (32,1024)
  gemm(zh,  if1_h, P(100), nullptr, h1h, 32, HID2, 2 * HID2, 4);  // silu
  gemm(h1h, if2_h, P(102), dout + Z_OFF, nullptr, 32, HID2, HID2, 0);
}
